// GroupedQueryAttention_46892452938096
// MI455X (gfx1250) — matrix operand order repaired and checked by reading, not compiled, not run on hardware
//
#include <hip/hip_runtime.h>
#include <hip/hip_bf16.h>

// ---------------------------------------------------------------------------
// GQA block for MI455X (gfx1250): bf16 WMMA GEMMs + flash attention.
// B=2, T=2048, C=2048, H=32, KV=8, HD=64.
// ---------------------------------------------------------------------------

typedef __attribute__((ext_vector_type(16))) __bf16 v16bf;
typedef __attribute__((ext_vector_type(8)))  __bf16 v8bf;
typedef __attribute__((ext_vector_type(8)))  float  v8f;

#define WMMA_BF16(a, b, c) \
  __builtin_amdgcn_wmma_f32_16x16x32_bf16(false, (a), false, (b), (short)0, (c), false, false)

// ----- fragment loaders (contiguous k order, same for A and B) -----------------
// A (16x32 bf16, MxK): lane<16 -> M=lane, K 0..15; lane>=16 -> K 16..31
__device__ __forceinline__ v16bf load_a_frag(const __bf16* A, int lda, int k0, int lane) {
  const int m    = lane & 15;
  const int koff = (lane >> 4) << 4;
  const __bf16* p = A + (size_t)m * lda + k0 + koff;
  v8bf lo = *(const v8bf*)p;
  v8bf hi = *(const v8bf*)(p + 8);
  v16bf a;
#pragma unroll
  for (int i = 0; i < 8; ++i) { a[i] = lo[i]; a[i + 8] = hi[i]; }
  return a;
}

// B (32x16 bf16, KxN) read from an N-major (row n contiguous in K) buffer:
// lane&15 = n, lane-half selects K 0..15 / 16..31 -> one contiguous 32B load.
__device__ __forceinline__ v16bf load_bT_frag(const __bf16* BT, int ldb, int n0, int k0, int lane) {
  const int n   = n0 + (lane & 15);
  const int klo = (lane >> 4) << 4;
  return *(const v16bf*)(BT + (size_t)n * ldb + k0 + klo);
}

// ----- DPP butterfly reductions over each 16-lane half ---------------------
// xor1 = quad_perm[1,0,3,2], xor2 = quad_perm[2,3,0,1],
// then ROW_HALF_MIRROR (lane^7) and ROW_MIRROR (lane^15) complete the tree.
template <int CTRL>
__device__ __forceinline__ float dpp_bfly(float x) {
  return __int_as_float(
      __builtin_amdgcn_update_dpp(0, __float_as_int(x), CTRL, 0xF, 0xF, true));
}
__device__ __forceinline__ float red_max16(float x) {
  x = fmaxf(x, dpp_bfly<0x0B1>(x));  // quad_perm [1,0,3,2]
  x = fmaxf(x, dpp_bfly<0x04E>(x));  // quad_perm [2,3,0,1]
  x = fmaxf(x, dpp_bfly<0x141>(x));  // row_half_mirror
  x = fmaxf(x, dpp_bfly<0x140>(x));  // row_mirror
  return x;
}
__device__ __forceinline__ float red_sum16(float x) {
  x += dpp_bfly<0x0B1>(x);
  x += dpp_bfly<0x04E>(x);
  x += dpp_bfly<0x141>(x);
  x += dpp_bfly<0x140>(x);
  return x;
}

// ----- elementwise / layout kernels ----------------------------------------
__global__ void cvt_f32_bf16(const float* __restrict__ in, __bf16* __restrict__ out, int n) {
  int i = blockIdx.x * blockDim.x + threadIdx.x;
  if (i < n) out[i] = (__bf16)in[i];
}

// in: (K x N) fp32 row-major -> out: (N x K) bf16 row-major
__global__ void transpose_cvt(const float* __restrict__ in, __bf16* __restrict__ out, int K, int N) {
  int i = blockIdx.x * blockDim.x + threadIdx.x;
  if (i >= N * K) return;
  int n = i / K;
  int k = i - n * K;
  out[i] = (__bf16)in[(size_t)k * N + n];
}

// RoPE + rearrange: src (B*T, H*64) bf16 -> dst (B, H, T, 64) bf16.
// `scale` folds the softmax 1/sqrt(hd) into Q.
__global__ void rope_rearrange(const __bf16* __restrict__ src, __bf16* __restrict__ dst,
                               int H, float scale) {
  int i = blockIdx.x * blockDim.x + threadIdx.x;
  const int W = H << 6;
  int row = i / W;            // b*T + t
  int col = i - row * W;      // h*64 + d
  int h = col >> 6, d = col & 63;
  int t = row & 2047, b = row >> 11;
  float cur   = (float)src[i];
  float other = (float)src[(size_t)row * W + (col ^ 32)];
  int j = d & 31;
  float inv = __powf(10000.0f, -(float)j * (1.0f / 32.0f));
  float ang = (float)t * inv;
  float s, c;
  __sincosf(ang, &s, &c);
  float val = (d < 32) ? (cur * c - other * s) : (cur * c + other * s);
  dst[(((size_t)(b * H + h)) * 2048 + t) * 64 + d] = (__bf16)(val * scale);
}

// V: src (B*T, 8*64) bf16 -> dst (B, KV, 64, T) bf16  (keys contiguous for P@V B-frags)
__global__ void transpose_v(const __bf16* __restrict__ src, __bf16* __restrict__ dst) {
  int i = blockIdx.x * blockDim.x + threadIdx.x;
  int row = i >> 9;           // / 512
  int col = i & 511;
  int kvh = col >> 6, d = col & 63;
  int t = row & 2047, b = row >> 11;
  dst[(((size_t)(b * 8 + kvh)) * 64 + d) * 2048 + t] = src[i];
}

// ----- bf16 WMMA GEMM: C(MxN) = A(MxK) @ BT(NxK)^T ------------------------
// 256 threads = 8 waves (2 M x 4 N); each wave computes a 32x64 tile
// (2 A-frags amortized over 4 shared B-frags -> 8 WMMAs per 6 frag loads).
template <typename OutT>
__global__ __launch_bounds__(256) void gemm_bf16(const __bf16* __restrict__ A,
                                                 const __bf16* __restrict__ BT,
                                                 OutT* __restrict__ C,
                                                 int M, int N, int K) {
  const int lane = threadIdx.x & 31;
  const int wid  = threadIdx.x >> 5;
  const int m0 = blockIdx.y * 64 + (wid >> 2) * 32;
  const int n0 = blockIdx.x * 256 + (wid & 3) * 64;

  v8f acc[2][4] = {};
  const __bf16* Ap0 = A + (size_t)m0 * K;
  const __bf16* Ap1 = A + (size_t)(m0 + 16) * K;

  for (int k0 = 0; k0 < K; k0 += 32) {
    if (k0 + 32 < K) {
      __builtin_prefetch(Ap0 + (size_t)(lane & 15) * K + k0 + 32, 0, 3);
      __builtin_prefetch(Ap1 + (size_t)(lane & 15) * K + k0 + 32, 0, 3);
    }
    v16bf a0 = load_a_frag(Ap0, K, k0, lane);
    v16bf a1 = load_a_frag(Ap1, K, k0, lane);
#pragma unroll
    for (int c = 0; c < 4; ++c) {
      v16bf b = load_bT_frag(BT, K, n0 + c * 16, k0, lane);
      acc[0][c] = WMMA_BF16(a0, b, acc[0][c]);
      acc[1][c] = WMMA_BF16(a1, b, acc[1][c]);
    }
  }

  const int half = lane >> 4, nl = lane & 15;
#pragma unroll
  for (int mi = 0; mi < 2; ++mi)
#pragma unroll
    for (int c = 0; c < 4; ++c)
#pragma unroll
      for (int r = 0; r < 8; ++r) {
        const int m = m0 + mi * 16 + r + (half << 3);
        const int n = n0 + c * 16 + nl;
        C[(size_t)m * N + n] = (OutT)acc[mi][c][r];
      }
}

// ----- flash attention (one wave per 16-query tile, 32 keys / iteration) ---
__global__ __launch_bounds__(256) void flash_attn(const __bf16* __restrict__ Qa,
                                                  const __bf16* __restrict__ Ka,
                                                  const __bf16* __restrict__ Va,
                                                  __bf16* __restrict__ Oa) {
  __shared__ __align__(64) __bf16 plds[8][16][32];
  const int lane = threadIdx.x & 31;
  const int wid  = threadIdx.x >> 5;
  const int bh = blockIdx.x;               // b*32 + h
  const int b = bh >> 5, h = bh & 31;
  const int kvh = h >> 2;                  // n_rep = 4
  const int qtile = blockIdx.y * 8 + wid;  // 0..127
  const int q0 = qtile << 4;
  const int half = lane >> 4, nl = lane & 15;

  const __bf16* Qp = Qa + ((size_t)bh * 2048 + q0) * 64;
  const __bf16* Kp = Ka + (size_t)(b * 8 + kvh) * 2048 * 64;
  const __bf16* Vp = Va + (size_t)(b * 8 + kvh) * 64 * 2048;

  const v16bf qf0 = load_a_frag(Qp, 64, 0, lane);
  const v16bf qf1 = load_a_frag(Qp, 64, 32, lane);

  v8f o[4] = {};
  float mrow[8], lrow[8];
#pragma unroll
  for (int r = 0; r < 8; ++r) { mrow[r] = -3.0e38f; lrow[r] = 0.0f; }

  const int nkt = (qtile + 2) >> 1;        // causal: keys up to q0+15
  for (int kt = 0; kt < nkt; ++kt) {
    const int kb = kt << 5;

    v8f s0 = {}, s1 = {};
    v16bf k00 = load_bT_frag(Kp, 64, kb, 0, lane);
    v16bf k01 = load_bT_frag(Kp, 64, kb, 32, lane);
    v16bf k10 = load_bT_frag(Kp, 64, kb + 16, 0, lane);
    v16bf k11 = load_bT_frag(Kp, 64, kb + 16, 32, lane);
    s0 = WMMA_BF16(qf0, k00, s0);
    s0 = WMMA_BF16(qf1, k01, s0);
    s1 = WMMA_BF16(qf0, k10, s1);
    s1 = WMMA_BF16(qf1, k11, s1);

#pragma unroll
    for (int r = 0; r < 8; ++r) {
      const int qrow = q0 + r + (half << 3);
      float v0 = (kb + nl      <= qrow) ? s0[r] : -3.0e38f;  // Q pre-scaled by 1/8
      float v1 = (kb + 16 + nl <= qrow) ? s1[r] : -3.0e38f;
      float tmax = red_max16(fmaxf(v0, v1));
      const float mnew  = fmaxf(mrow[r], tmax);
      const float alpha = __expf(mrow[r] - mnew);
      const float p0 = __expf(v0 - mnew);
      const float p1 = __expf(v1 - mnew);
      const float rs = red_sum16(p0 + p1);
      lrow[r] = lrow[r] * alpha + rs;
      mrow[r] = mnew;
#pragma unroll
      for (int c = 0; c < 4; ++c) o[c][r] *= alpha;
      plds[wid][r + (half << 3)][nl]      = (__bf16)p0;
      plds[wid][r + (half << 3)][nl + 16] = (__bf16)p1;
    }

    // transpose P (C-layout -> A-layout) through this wave's LDS slice
    __builtin_amdgcn_wave_barrier();
    asm volatile("s_wait_dscnt 0" ::: "memory");
    const __bf16* pr = &plds[wid][nl][0];
    v8bf plo = *(const v8bf*)(pr + (half << 4));
    v8bf phi = *(const v8bf*)(pr + 8 + (half << 4));
    v16bf pf;
#pragma unroll
    for (int i = 0; i < 8; ++i) { pf[i] = plo[i]; pf[i + 8] = phi[i]; }
    __builtin_amdgcn_wave_barrier();

#pragma unroll
    for (int c = 0; c < 4; ++c) {
      v16bf vf = load_bT_frag(Vp, 2048, c * 16, kb, lane);  // Vt row = d, contiguous keys
      o[c] = WMMA_BF16(pf, vf, o[c]);
    }
  }

#pragma unroll
  for (int c = 0; c < 4; ++c)
#pragma unroll
    for (int r = 0; r < 8; ++r) {
      const int row = b * 2048 + q0 + r + (half << 3);
      const int col = h * 64 + c * 16 + nl;
      Oa[(size_t)row * 2048 + col] = (__bf16)(o[c][r] / lrow[r]);
    }
}

// ----- workspace layout (bytes, all offsets 256B-aligned) ------------------
static constexpr size_t OFF_XB  = 0;                               // x bf16     (4096x2048)
static constexpr size_t OFF_WQT = OFF_XB  + 4096ull * 2048 * 2;    // WqT bf16   (2048x2048)
static constexpr size_t OFF_WKT = OFF_WQT + 2048ull * 2048 * 2;    // WkT bf16   (512x2048)
static constexpr size_t OFF_WVT = OFF_WKT + 512ull  * 2048 * 2;    // WvT bf16   (512x2048)
static constexpr size_t OFF_WOT = OFF_WVT + 512ull  * 2048 * 2;    // WoT bf16   (2048x2048)
static constexpr size_t OFF_QG  = OFF_WOT + 2048ull * 2048 * 2;    // Q gemm out (4096x2048)
static constexpr size_t OFF_KG  = OFF_QG  + 4096ull * 2048 * 2;    // K gemm out (4096x512)
static constexpr size_t OFF_VG  = OFF_KG  + 4096ull * 512 * 2;     // V gemm out (4096x512)
static constexpr size_t OFF_QA  = OFF_VG  + 4096ull * 512 * 2;     // Q rope     (B,H,T,64)
static constexpr size_t OFF_KA  = OFF_QA  + 4096ull * 2048 * 2;    // K rope     (B,KV,T,64)
static constexpr size_t OFF_VA  = OFF_KA  + 4096ull * 512 * 2;     // V^T        (B,KV,64,T)
static constexpr size_t OFF_OA  = OFF_VA  + 4096ull * 512 * 2;     // attn out   (4096x2048)

extern "C" void kernel_launch(void* const* d_in, const int* in_sizes, int n_in,
                              void* d_out, int out_size, void* d_ws, size_t ws_size,
                              hipStream_t stream) {
  (void)in_sizes; (void)n_in; (void)out_size; (void)ws_size;
  const float* x  = (const float*)d_in[0];
  const float* Wq = (const float*)d_in[1];
  const float* Wk = (const float*)d_in[2];
  const float* Wv = (const float*)d_in[3];
  const float* Wo = (const float*)d_in[4];
  float* out = (float*)d_out;

  char* ws = (char*)d_ws;
  __bf16* xb  = (__bf16*)(ws + OFF_XB);
  __bf16* wqT = (__bf16*)(ws + OFF_WQT);
  __bf16* wkT = (__bf16*)(ws + OFF_WKT);
  __bf16* wvT = (__bf16*)(ws + OFF_WVT);
  __bf16* woT = (__bf16*)(ws + OFF_WOT);
  __bf16* qg  = (__bf16*)(ws + OFF_QG);
  __bf16* kg  = (__bf16*)(ws + OFF_KG);
  __bf16* vg  = (__bf16*)(ws + OFF_VG);
  __bf16* qa  = (__bf16*)(ws + OFF_QA);
  __bf16* ka  = (__bf16*)(ws + OFF_KA);
  __bf16* va  = (__bf16*)(ws + OFF_VA);
  __bf16* oa  = (__bf16*)(ws + OFF_OA);

  const int M = 4096;  // B*T

  // 1. convert activations + weights (weights transposed for WMMA B-frags)
  cvt_f32_bf16<<<(M * 2048) / 256, 256, 0, stream>>>(x, xb, M * 2048);
  transpose_cvt<<<(2048 * 2048) / 256, 256, 0, stream>>>(Wq, wqT, 2048, 2048);
  transpose_cvt<<<(2048 * 512) / 256, 256, 0, stream>>>(Wk, wkT, 2048, 512);
  transpose_cvt<<<(2048 * 512) / 256, 256, 0, stream>>>(Wv, wvT, 2048, 512);
  transpose_cvt<<<(2048 * 2048) / 256, 256, 0, stream>>>(Wo, woT, 2048, 2048);

  // 2. QKV projections (bf16 WMMA, fp32 accumulate, bf16 out)
  gemm_bf16<__bf16><<<dim3(2048 / 256, M / 64), 256, 0, stream>>>(xb, wqT, qg, M, 2048, 2048);
  gemm_bf16<__bf16><<<dim3(512 / 256,  M / 64), 256, 0, stream>>>(xb, wkT, kg, M, 512, 2048);
  gemm_bf16<__bf16><<<dim3(512 / 256,  M / 64), 256, 0, stream>>>(xb, wvT, vg, M, 512, 2048);

  // 3. RoPE + attention-friendly layouts (Q pre-scaled by 1/sqrt(64))
  rope_rearrange<<<(M * 2048) / 256, 256, 0, stream>>>(qg, qa, 32, 0.125f);
  rope_rearrange<<<(M * 512) / 256, 256, 0, stream>>>(kg, ka, 8, 1.0f);
  transpose_v<<<(M * 512) / 256, 256, 0, stream>>>(vg, va);

  // 4. causal flash attention (64 (b,h) pairs x 128 query tiles / 8 waves)
  flash_attn<<<dim3(64, 16), 256, 0, stream>>>(qa, ka, va, oa);

  // 5. output projection -> fp32
  gemm_bf16<float><<<dim3(2048 / 256, M / 64), 256, 0, stream>>>(oa, woT, out, M, 2048, 2048);
}
